// MultiheadAttention_82532091560212
// MI455X (gfx1250) — matrix operand order repaired and checked by reading, not compiled, not run on hardware
//
#include <hip/hip_runtime.h>
#include <hip/hip_bf16.h>

// ---------------------------------------------------------------------------
// Multi-head attention forward for MI455X (gfx1250, wave32, WMMA + TDM).
// B=2, T=2048, E=1024, H=16, D=64.
//   q = X Wq^T ; k = X Wk^T ; v = X Wv^T        (WMMA f16 GEMM, f32 acc)
//   flash-attention core (causal, online softmax, WMMA; Q/K tiles via
//   TENSOR_LOAD_TO_LDS with LDS row padding)
//   out = attn Wo^T                              (WMMA f16 GEMM, TDM A tiles)
// Intermediates kept in f16 workspace (4 x 8 MiB).
// ---------------------------------------------------------------------------

typedef __attribute__((ext_vector_type(16))) _Float16 v16h;
typedef __attribute__((ext_vector_type(8)))  _Float16 v8h;
typedef __attribute__((ext_vector_type(8)))  float    v8f;
typedef __attribute__((ext_vector_type(4)))  unsigned int u32x4;
typedef __attribute__((ext_vector_type(8)))  int i32x8;
typedef __attribute__((ext_vector_type(4)))  int i32x4;

#define BB 2
#define TT 2048
#define EE 1024
#define HH 16
#define DD 64
#define MM (BB * TT)   // 4096 rows

// ---------------------------------------------------------------------------
// Tensor Data Mover: load a rows x width(halves) f16 tile from a row-major
// tensor with row stride EE halves into LDS with 72-half padded rows.
// Pad: tile row = 64 halves = 32 DWORDs -> pad_interval code 4 (32 DW),
// pad_amount code 3 (4 DWORDs = 8 halves) => 72-half LDS row stride.
// Issue from one wave; tracked by TENSORcnt.
// ---------------------------------------------------------------------------
__device__ __forceinline__ void tdm_load_tile_f16(const _Float16* gsrc,
                                                  unsigned lds_off,
                                                  int width_halves, int rows) {
  const unsigned long long ga = (unsigned long long)(const void*)gsrc;
  u32x4 g0;
  g0[0] = 1u;                                   // count=1, user-mode load
  g0[1] = lds_off;                              // lds_addr (bytes)
  g0[2] = (unsigned)(ga & 0xffffffffu);         // global_addr[31:0]
  g0[3] = (unsigned)((ga >> 32) & 0x01ffffffu)  // global_addr[56:32]
        | (2u << 30);                           // type = 2 ("image")
  i32x8 g1;
  g1[0] = (1 << 16)        // data_size = 2 bytes
        | (1 << 20)        // pad_enable
        | (4 << 22)        // pad_interval: every 32 DWORDs
        | (3 << 25);       // pad_amount: 4 DWORDs
  g1[1] = (width_halves & 0xffff) << 16;        // tensor_dim0[15:0]
  g1[2] = (rows & 0xffff) << 16;                // tensor_dim1[15:0]
  g1[3] = (width_halves & 0xffff) << 16;        // tile_dim0
  g1[4] = rows & 0xffff;                        // tile_dim1 (tile_dim2 = 0)
  g1[5] = EE;                                   // tensor_dim0_stride[31:0]
  g1[6] = 0;
  g1[7] = 0;
  i32x4 z4 = {0, 0, 0, 0};                      // 2D: groups 2/3 unused
#if __clang_major__ >= 23
  i32x8 z8 = {0, 0, 0, 0, 0, 0, 0, 0};
  __builtin_amdgcn_tensor_load_to_lds(g0, g1, z4, z4, z8, 0);
#else
  __builtin_amdgcn_tensor_load_to_lds(g0, g1, z4, z4, 0);
#endif
}

// ---------------------------------------------------------------------------
// WMMA fragment helpers (wave32, V_WMMA_F32_16X16X32_F16 layouts per ISA 7.12.2)
// ---------------------------------------------------------------------------
__device__ __forceinline__ v16h frag_cat(v8h lo, v8h hi) {
  v16h r;
#pragma unroll
  for (int i = 0; i < 8; ++i) { r[i] = lo[i]; r[i + 8] = hi[i]; }
  return r;
}

// A matrix 16x32 f16. lanes 0-15: row M=lane, K={kb+0..7, kb+16..23};
// lanes 16-31: row M=lane-16, K={kb+8..15, kb+24..31}. Two ds_load_b128.
__device__ __forceinline__ v16h load_fragA(const _Float16* lds, int row_base,
                                           int stride, int kbase, int lane) {
  const int m  = lane & 15;
  const int ko = kbase + ((lane >> 4) << 3);
  const _Float16* p = lds + (row_base + m) * stride + ko;
  v8h lo = *(const v8h*)(p);
  v8h hi = *(const v8h*)(p + 16);
  return frag_cat(lo, hi);
}

// B matrix 32x16 f16 stored n-major in LDS (Bs[n][k]). lanes 0-15: col N=lane,
// K={kb+0..7, kb+16..23}; lanes 16-31: col N=lane-16, K={kb+8..15, kb+24..31}.
__device__ __forceinline__ v16h load_fragB(const _Float16* lds, int n_base,
                                           int stride, int kbase, int lane) {
  const int n  = lane & 15;
  const int ko = kbase + ((lane >> 4) << 3);
  const _Float16* p = lds + (n_base + n) * stride + ko;
  v8h lo = *(const v8h*)(p);
  v8h hi = *(const v8h*)(p + 16);
  return frag_cat(lo, hi);
}

__device__ __forceinline__ v8f wmma_f16(v16h a, v16h b, v8f c) {
  return __builtin_amdgcn_wmma_f32_16x16x32_f16(
      /*neg_a=*/false, a, /*neg_b=*/false, b,
      /*c_mod=*/(short)0, c, /*reuse_a=*/false, /*reuse_b=*/false);
}

// ---------------------------------------------------------------------------
// Tile staging: global -> LDS f16
// ---------------------------------------------------------------------------
// 128x64 A tile from f32 source, 8 float4 per thread (256 threads), convert.
__device__ __forceinline__ void stage_A(const float* X, _Float16* As, int m_blk,
                                        int K, int k0, int tid) {
#pragma unroll
  for (int i = 0; i < 8; ++i) {
    const int idx = tid + i * 256;
    const int r = idx >> 4, c = (idx & 15) << 2;
    const float4 v = *(const float4*)(X + (size_t)(m_blk + r) * K + k0 + c);
    _Float16* d = As + r * 72 + c;
    d[0] = (_Float16)v.x; d[1] = (_Float16)v.y;
    d[2] = (_Float16)v.z; d[3] = (_Float16)v.w;
  }
}

// 128x64 A tile from f16 source: one TDM DMA issued by wave 0.
__device__ __forceinline__ void stage_A(const _Float16* X, _Float16* As,
                                        int m_blk, int K, int k0, int tid) {
  if (tid < 32) {
    tdm_load_tile_f16(X + (size_t)m_blk * K + k0,
                      (unsigned)(uintptr_t)(void*)As, 64, 128);
  }
}

// ---------------------------------------------------------------------------
// GEMM: Y[M,N] = X[M,K] * W[N,K]^T   (W row-major f32; X f32 or f16; Y f16/f32)
// Block 256 thr (8 waves); tile 128x64, BK=64; wave -> 32x32 (2x2 WMMA tiles).
// ---------------------------------------------------------------------------
template <typename AT, typename OT>
__global__ void __launch_bounds__(256)
mha_gemm_xwT(const AT* __restrict__ X, const float* __restrict__ W,
             OT* __restrict__ Y, int M, int N, int K) {
  constexpr int BM = 128, BN = 64, BK = 64, ST = 72;
  __shared__ __align__(16) _Float16 As[BM * ST];
  __shared__ __align__(16) _Float16 Bs[BN * ST];

  const int tid  = threadIdx.x;
  const int lane = tid & 31;
  const int wid  = tid >> 5;
  const int m_blk = blockIdx.y * BM;
  const int n_blk = blockIdx.x * BN;
  const int wm = (wid & 3) * 32;   // wave row offset in tile
  const int wn = (wid >> 2) * 32;  // wave col offset in tile

  v8f acc[2][2] = {};

  for (int k0 = 0; k0 < K; k0 += BK) {
    stage_A(X, As, m_blk, K, k0, tid);
    // B tile: 64 rows of W (f32), 64 cols -> 4 float4 per thread, convert.
#pragma unroll
    for (int i = 0; i < 4; ++i) {
      const int idx = tid + i * 256;
      const int r = idx >> 4, c = (idx & 15) << 2;
      const float4 v = *(const float4*)(W + (size_t)(n_blk + r) * K + k0 + c);
      _Float16* d = Bs + r * ST + c;
      d[0] = (_Float16)v.x; d[1] = (_Float16)v.y;
      d[2] = (_Float16)v.z; d[3] = (_Float16)v.w;
    }
    // Prefetch next K slab of X (global_prefetch_b8).
    if (k0 + BK < K) {
      __builtin_prefetch(X + (size_t)(m_blk + (tid >> 1)) * K + k0 + BK +
                             (tid & 1) * 32, 0, 0);
    }
    if (tid < 32) __builtin_amdgcn_s_wait_tensorcnt(0);  // no-op in f32 path
    __syncthreads();

#pragma unroll
    for (int kc = 0; kc < BK; kc += 32) {
      v16h a0 = load_fragA(As, wm,      ST, kc, lane);
      v16h a1 = load_fragA(As, wm + 16, ST, kc, lane);
      v16h b0 = load_fragB(Bs, wn,      ST, kc, lane);
      v16h b1 = load_fragB(Bs, wn + 16, ST, kc, lane);
      acc[0][0] = wmma_f16(a0, b0, acc[0][0]);
      acc[0][1] = wmma_f16(a0, b1, acc[0][1]);
      acc[1][0] = wmma_f16(a1, b0, acc[1][0]);
      acc[1][1] = wmma_f16(a1, b1, acc[1][1]);
    }
    __syncthreads();
  }

  // C layout: lanes 0-15 col N=lane rows 0..7; lanes 16-31 col N=lane-16 rows 8..15.
  const int nN = lane & 15;
  const int mo = (lane >> 4) << 3;
#pragma unroll
  for (int im = 0; im < 2; ++im) {
#pragma unroll
    for (int jn = 0; jn < 2; ++jn) {
      const size_t row0 = (size_t)(m_blk + wm + im * 16 + mo);
      const int col = n_blk + wn + jn * 16 + nN;
#pragma unroll
      for (int r = 0; r < 8; ++r)
        Y[(row0 + r) * N + col] = (OT)acc[im][jn][r];
    }
  }
}

// ---------------------------------------------------------------------------
// Flash attention core (causal). Block = (b, h, 128-query tile); 8 waves;
// each wave owns 16 query rows x full D=64. KV tiles of 64 keys.
// Q and K tiles staged by the Tensor Data Mover; V transposed via VALU.
// ---------------------------------------------------------------------------
__global__ void __launch_bounds__(256)
mha_flash_attn(const _Float16* __restrict__ qh, const _Float16* __restrict__ kh,
               const _Float16* __restrict__ vh, _Float16* __restrict__ ah) {
  constexpr int TQ = 128, TK = 64, ST = 72;
  __shared__ __align__(16) _Float16 Qs [TQ * ST];
  __shared__ __align__(16) _Float16 Ks [TK * ST];
  __shared__ __align__(16) _Float16 Vts[DD * ST];   // transposed: Vts[d][key]
  __shared__ __align__(16) _Float16 Ps [TQ * ST];

  const int tid  = threadIdx.x;
  const int lane = tid & 31;
  const int wid  = tid >> 5;
  const int qt = blockIdx.x;
  const int b  = blockIdx.y / HH;
  const int h  = blockIdx.y % HH;

  const size_t base = ((size_t)b * TT) * EE + (size_t)h * DD;
  const _Float16* qp = qh + base;
  const _Float16* kp = kh + base;
  const _Float16* vp = vh + base;
  _Float16*       ap = ah + base;
  const int q0 = qt * TQ;

  // Q tile (128x64 halves) via TDM, issued once by wave 0.
  if (tid < 32) {
    tdm_load_tile_f16(qp + (size_t)q0 * EE, (unsigned)(uintptr_t)(void*)Qs,
                      DD, TQ);
  }

  const int nN = lane & 15;
  const int mo = (lane >> 4) << 3;
  const int qrow = q0 + wid * 16 + mo;  // +r gives this lane's query rows

  float mprev[8], lsum[8];
  v8f o[4] = {};
#pragma unroll
  for (int r = 0; r < 8; ++r) { mprev[r] = -1e30f; lsum[r] = 0.f; }

  const int kv_end = q0 + TQ;  // causal: keys beyond query tile never needed
  for (int kv0 = 0; kv0 < kv_end; kv0 += TK) {
    __syncthreads();  // previous iteration finished with Ks/Vts
    // K tile 64x64 halves via TDM (overlaps with V transpose staging below).
    if (tid < 32) {
      tdm_load_tile_f16(kp + (size_t)kv0 * EE, (unsigned)(uintptr_t)(void*)Ks,
                        DD, TK);
    }
    // V tile transposed into Vts[d][key] (2 v8h per thread + 16 b16 scatters).
#pragma unroll
    for (int i = 0; i < 2; ++i) {
      const int idx = tid + i * 256;
      const int r = idx >> 3, c = (idx & 7) << 3;
      v8h v = *(const v8h*)(vp + (size_t)(kv0 + r) * EE + c);
#pragma unroll
      for (int e = 0; e < 8; ++e) Vts[(c + e) * ST + r] = v[e];
    }
    if (tid < 32) __builtin_amdgcn_s_wait_tensorcnt(0);  // Q (+K) DMA done
    __syncthreads();

    // S = Q K^T (16x64 per wave -> 4 WMMA tiles, K-dim = D = 2 chunks)
    v8f s[4] = {};
#pragma unroll
    for (int kc = 0; kc < DD; kc += 32) {
      v16h a = load_fragA(Qs, wid * 16, ST, kc, lane);
#pragma unroll
      for (int jt = 0; jt < 4; ++jt) {
        v16h bf = load_fragB(Ks, jt * 16, ST, kc, lane);
        s[jt] = wmma_f16(a, bf, s[jt]);
      }
    }

    // Scale + causal mask + online softmax. Row stats reduce over 16-lane
    // halves (matches C layout: rows 0..7 in lanes 0..15, rows 8..15 above).
    float tmax[8];
#pragma unroll
    for (int r = 0; r < 8; ++r) tmax[r] = -1e30f;
#pragma unroll
    for (int jt = 0; jt < 4; ++jt) {
      const int key = kv0 + jt * 16 + nN;
#pragma unroll
      for (int r = 0; r < 8; ++r) {
        float sv = s[jt][r] * 0.125f;           // 1/sqrt(64)
        if (key > qrow + r) sv = -1e30f;        // causal mask
        s[jt][r] = sv;
        tmax[r] = fmaxf(tmax[r], sv);
      }
    }
#pragma unroll
    for (int r = 0; r < 8; ++r) {
      float t = tmax[r];
      t = fmaxf(t, __shfl_xor(t, 1));
      t = fmaxf(t, __shfl_xor(t, 2));
      t = fmaxf(t, __shfl_xor(t, 4));
      t = fmaxf(t, __shfl_xor(t, 8));
      const float mn = fmaxf(mprev[r], t);
      const float alpha = __expf(mprev[r] - mn);
      mprev[r] = mn;
      float rs = 0.f;
#pragma unroll
      for (int jt = 0; jt < 4; ++jt) {
        const float p = __expf(s[jt][r] - mn);
        s[jt][r] = p;
        rs += p;
      }
      rs += __shfl_xor(rs, 1);
      rs += __shfl_xor(rs, 2);
      rs += __shfl_xor(rs, 4);
      rs += __shfl_xor(rs, 8);
      lsum[r] = lsum[r] * alpha + rs;
#pragma unroll
      for (int jt = 0; jt < 4; ++jt) o[jt][r] *= alpha;
    }

    // P -> LDS (per-wave private rows; same-wave DS ordering suffices).
#pragma unroll
    for (int jt = 0; jt < 4; ++jt)
#pragma unroll
      for (int r = 0; r < 8; ++r)
        Ps[(wid * 16 + mo + r) * ST + jt * 16 + nN] = (_Float16)s[jt][r];

    // O += P V  (A = P 16x64 over key chunks, B = Vts)
#pragma unroll
    for (int kc = 0; kc < TK; kc += 32) {
      v16h a = load_fragA(Ps, wid * 16, ST, kc, lane);
#pragma unroll
      for (int jt = 0; jt < 4; ++jt) {
        v16h bf = load_fragB(Vts, jt * 16, ST, kc, lane);
        o[jt] = wmma_f16(a, bf, o[jt]);
      }
    }
  }

  // Normalize and store attn output [b][t][h*D + d] as f16.
#pragma unroll
  for (int r = 0; r < 8; ++r) {
    const float inv = 1.0f / lsum[r];
#pragma unroll
    for (int jt = 0; jt < 4; ++jt)
      ap[(size_t)(qrow + r) * EE + jt * 16 + nN] = (_Float16)(o[jt][r] * inv);
  }
}

// ---------------------------------------------------------------------------
// Launch: 3 projection GEMMs -> flash attention -> output projection GEMM.
// Inputs: 0=queries 1=keys 2=values 3=mask(unused; causal computed) 4..7=Wq..Wo
// ---------------------------------------------------------------------------
extern "C" void kernel_launch(void* const* d_in, const int* in_sizes, int n_in,
                              void* d_out, int out_size, void* d_ws,
                              size_t ws_size, hipStream_t stream) {
  const float* q  = (const float*)d_in[0];
  const float* k  = (const float*)d_in[1];
  const float* v  = (const float*)d_in[2];
  const float* Wq = (const float*)d_in[4];
  const float* Wk = (const float*)d_in[5];
  const float* Wv = (const float*)d_in[6];
  const float* Wo = (const float*)d_in[7];
  float* out = (float*)d_out;

  const size_t plane = (size_t)MM * EE;  // 4096*1024 halves
  _Float16* qh = (_Float16*)d_ws;
  _Float16* kh = qh + plane;
  _Float16* vh = kh + plane;
  _Float16* ah = vh + plane;

  dim3 blk(256);
  dim3 gg(EE / 64, MM / 128);   // (16, 32)
  mha_gemm_xwT<float, _Float16><<<gg, blk, 0, stream>>>(q, Wq, qh, MM, EE, EE);
  mha_gemm_xwT<float, _Float16><<<gg, blk, 0, stream>>>(k, Wk, kh, MM, EE, EE);
  mha_gemm_xwT<float, _Float16><<<gg, blk, 0, stream>>>(v, Wv, vh, MM, EE, EE);

  dim3 ga(TT / 128, BB * HH);   // (16, 32)
  mha_flash_attn<<<ga, blk, 0, stream>>>(qh, kh, vh, ah);

  mha_gemm_xwT<_Float16, float><<<gg, blk, 0, stream>>>(ah, Wo, out, MM, EE, EE);
}
